// Model_10514079941130
// MI455X (gfx1250) — compile-verified, matrix operand order checked by reading, not run on hardware
//
#include <hip/hip_runtime.h>

typedef __attribute__((ext_vector_type(16))) _Float16 v16h;
typedef __attribute__((ext_vector_type(8)))  _Float16 v8h;
typedef __attribute__((ext_vector_type(4)))  _Float16 v4h;
typedef __attribute__((ext_vector_type(8)))  float    v8f;

#define NLAYER 12
#define DMODEL 768
#define NHEAD  12
#define HD     64
#define FFDIM  3072
#define SEQLEN 197
#define SEQ2   208                 // 13 * 16, K/V LDS rows
#define VTP    216                 // transposed-V LDS row stride (f16), 432B = 16B aligned
#define NBATCH 64
#define NTOK   (NBATCH * SEQLEN)   // 12608
#define NPATCH 196
#define PROWS  (NBATCH * NPATCH)   // 12544

// ---------------------------------------------------------------------------
// block-wide sum reduction (256 threads, wave32)
// ---------------------------------------------------------------------------
__device__ __forceinline__ float block_sum256(float v, float* red) {
#pragma unroll
    for (int o = 16; o > 0; o >>= 1) v += __shfl_xor(v, o, 32);
    __syncthreads();
    if ((threadIdx.x & 31) == 0) red[threadIdx.x >> 5] = v;
    __syncthreads();
    float t = 0.f;
#pragma unroll
    for (int i = 0; i < 8; ++i) t += red[i];
    return t;
}

// ---------------------------------------------------------------------------
// im2col: image [B,3,224,224] -> H [12544, 768]  (k = c*256 + i*16 + j)
// ---------------------------------------------------------------------------
__global__ void im2col_kernel(const float* __restrict__ img, float* __restrict__ out) {
    size_t total = (size_t)PROWS * DMODEL;
    for (size_t idx = (size_t)blockIdx.x * blockDim.x + threadIdx.x; idx < total;
         idx += (size_t)gridDim.x * blockDim.x) {
        int k = (int)(idx % DMODEL);
        size_t row = idx / DMODEL;
        int p = (int)(row % NPATCH);
        int b = (int)(row / NPATCH);
        int c = k >> 8, rem = k & 255, i = rem >> 4, j = rem & 15;
        int pi = p / 14, pj = p % 14;
        int y = pi * 16 + i, x = pj * 16 + j;
        out[idx] = img[(((size_t)b * 3 + c) * 224 + y) * 224 + x];
    }
}

// ---------------------------------------------------------------------------
// assemble tokens (cls + patches + pos_emb) then ln_pre  -> X [12608, 768]
// ---------------------------------------------------------------------------
__global__ __launch_bounds__(256)
void assemble_ln_kernel(const float* __restrict__ patches, const float* __restrict__ cls,
                        const float* __restrict__ pos, const float* __restrict__ g,
                        const float* __restrict__ bb, float* __restrict__ out) {
    __shared__ float red[8];
    int row = blockIdx.x;
    int b = row / SEQLEN, s = row % SEQLEN;
    int tid = threadIdx.x;
    const float* src = (s == 0) ? cls : (patches + ((size_t)b * NPATCH + (s - 1)) * DMODEL);
    float v[3]; float sum = 0.f;
#pragma unroll
    for (int q = 0; q < 3; ++q) {
        int d = tid + q * 256;
        float x = src[d] + pos[(size_t)s * DMODEL + d];
        v[q] = x; sum += x;
    }
    float mu = block_sum256(sum, red) * (1.f / DMODEL);
    float s2 = 0.f;
#pragma unroll
    for (int q = 0; q < 3; ++q) { float d = v[q] - mu; s2 += d * d; }
    float var = block_sum256(s2, red) * (1.f / DMODEL);
    float rstd = rsqrtf(var + 1e-5f);
    float* dst = out + (size_t)row * DMODEL;
#pragma unroll
    for (int q = 0; q < 3; ++q) {
        int d = tid + q * 256;
        dst[d] = (v[q] - mu) * rstd * g[d] + bb[d];
    }
}

// ---------------------------------------------------------------------------
// layernorm: out[row,:] = LN(in[row*rstride : +768]) * g + b
// ---------------------------------------------------------------------------
__global__ __launch_bounds__(256)
void ln_kernel(const float* __restrict__ in, const float* __restrict__ g,
               const float* __restrict__ bb, float* __restrict__ out, int rstride) {
    __shared__ float red[8];
    int row = blockIdx.x, tid = threadIdx.x;
    const float* src = in + (size_t)row * rstride;
    float v[3]; float sum = 0.f;
#pragma unroll
    for (int q = 0; q < 3; ++q) { v[q] = src[tid + q * 256]; sum += v[q]; }
    float mu = block_sum256(sum, red) * (1.f / DMODEL);
    float s2 = 0.f;
#pragma unroll
    for (int q = 0; q < 3; ++q) { float d = v[q] - mu; s2 += d * d; }
    float var = block_sum256(s2, red) * (1.f / DMODEL);
    float rstd = rsqrtf(var + 1e-5f);
    float* dst = out + (size_t)row * DMODEL;
#pragma unroll
    for (int q = 0; q < 3; ++q) {
        int d = tid + q * 256;
        dst[d] = (v[q] - mu) * rstd * g[d] + bb[d];
    }
}

// build a 16-element A/B fragment from two contiguous 8-f16 groups
__device__ __forceinline__ v16h frag_from_groups(v8h g0, v8h g1) {
    v16h f;
#pragma unroll
    for (int e = 0; e < 8; ++e) { f[e] = g0[e]; f[8 + e] = g1[e]; }
    return f;
}

// ---------------------------------------------------------------------------
// WMMA GEMM:  C[M,N] = A[M,K] (f32 row-major) x W[N,K]^T (f32 row-major) + bias
// EPI: 0 = store, 1 = store + residual(Res), 2 = store + quickgelu
// tile 128x128x64, 8 waves (4 M x 2 N), each wave 2x4 fragments of 16x16x32 f16
// ---------------------------------------------------------------------------
template <int EPI>
__global__ __launch_bounds__(256)
void gemm_kernel(const float* __restrict__ A, int lda,
                 const float* __restrict__ W, int ldw,
                 const float* __restrict__ bias,
                 const float* __restrict__ Res,
                 float* __restrict__ C, int ldc,
                 int M, int N, int K) {
    __shared__ _Float16 As[128 * 72];
    __shared__ _Float16 Ws[128 * 72];
    const int tid = threadIdx.x;
    const int lane = tid & 31, wid = tid >> 5;
    const int waveM = wid & 3, waveN = wid >> 2;  // 4 x 2 wave grid
    const int m0 = blockIdx.y * 128, n0 = blockIdx.x * 128;
    const int halfid = lane >> 4, col = lane & 15;

    v8f acc[2][4];
#pragma unroll
    for (int a = 0; a < 2; ++a)
#pragma unroll
        for (int bq = 0; bq < 4; ++bq) acc[a][bq] = v8f{};

    const int nk = K >> 6;  // K always a multiple of 64 here
    for (int kt = 0; kt < nk; ++kt) {
        const int k0 = kt << 6;
        // stage 128x64 f32 -> f16 LDS tiles, float4 global loads
#pragma unroll
        for (int i = 0; i < 8; ++i) {
            int idx = (i * 256 + tid) << 2;   // element index 0..8191
            int r = idx >> 6, c = idx & 63;
            int gm = m0 + r;
            float4 va = make_float4(0.f, 0.f, 0.f, 0.f);
            if (gm < M) va = *(const float4*)&A[(size_t)gm * lda + k0 + c];
            v4h ha;
            ha[0] = (_Float16)va.x; ha[1] = (_Float16)va.y;
            ha[2] = (_Float16)va.z; ha[3] = (_Float16)va.w;
            *(v4h*)&As[r * 72 + c] = ha;
            int gn = n0 + r;
            float4 vw = make_float4(0.f, 0.f, 0.f, 0.f);
            if (gn < N) vw = *(const float4*)&W[(size_t)gn * ldw + k0 + c];
            v4h hw;
            hw[0] = (_Float16)vw.x; hw[1] = (_Float16)vw.y;
            hw[2] = (_Float16)vw.z; hw[3] = (_Float16)vw.w;
            *(v4h*)&Ws[r * 72 + c] = hw;
        }
        // prefetch next K-tile into caches (global_prefetch_b8)
        if (kt + 1 < nk) {
            int r = tid & 127;
            int ch = (tid >> 7) << 5;  // 0 / 32 -> second cacheline of the row
            int gm = m0 + r;
            if (gm < M) __builtin_prefetch(&A[(size_t)gm * lda + k0 + 64 + ch], 0, 1);
            int gn = n0 + r;
            if (gn < N) __builtin_prefetch(&W[(size_t)gn * ldw + k0 + 64 + ch], 0, 1);
        }
        __syncthreads();
#pragma unroll
        for (int kb = 0; kb < 64; kb += 32) {
            v16h af[2], bf[4];
#pragma unroll
            for (int mf = 0; mf < 2; ++mf) {
                int rm = waveM * 32 + mf * 16 + col;
                const _Float16* ap = &As[rm * 72 + kb + (halfid << 3)];
                af[mf] = frag_from_groups(*(const v8h*)ap, *(const v8h*)&ap[16]);
            }
#pragma unroll
            for (int nf = 0; nf < 4; ++nf) {
                int rn = waveN * 64 + nf * 16 + col;
                const _Float16* wp = &Ws[rn * 72 + kb + (halfid << 3)];
                bf[nf] = frag_from_groups(*(const v8h*)wp, *(const v8h*)&wp[16]);
            }
#pragma unroll
            for (int mf = 0; mf < 2; ++mf)
#pragma unroll
                for (int nf = 0; nf < 4; ++nf)
                    acc[mf][nf] = __builtin_amdgcn_wmma_f32_16x16x32_f16(
                        false, af[mf], false, bf[nf], (short)0, acc[mf][nf], false, false);
        }
        __syncthreads();
    }

#pragma unroll
    for (int mf = 0; mf < 2; ++mf)
#pragma unroll
        for (int nf = 0; nf < 4; ++nf)
#pragma unroll
            for (int r = 0; r < 8; ++r) {
                int m = m0 + waveM * 32 + mf * 16 + r + halfid * 8;
                int n = n0 + waveN * 64 + nf * 16 + col;
                if (m < M && n < N) {
                    float v = acc[mf][nf][r];
                    if (bias) v += bias[n];
                    if (EPI == 1) v += Res[(size_t)m * ldc + n];
                    if (EPI == 2) v = v / (1.f + __expf(-1.702f * v));
                    C[(size_t)m * ldc + n] = v;
                }
            }
}

// ---------------------------------------------------------------------------
// flash-style attention: one block (8 waves) per (b, head).
// QKV packed f32 [12608, 2304]; output O f32 [12608, 768].
// K row-major + V TRANSPOSED in LDS (f16) so every fragment half is a
// contiguous 8xf16 group -> ds_load_b128. Per-wave P transpose buffer.
// ---------------------------------------------------------------------------
__global__ __launch_bounds__(256)
void attn_kernel(const float* __restrict__ qkv, float* __restrict__ out) {
    __shared__ _Float16 Klds[SEQ2 * HD];        // 26 KB, [seq][hd]
    __shared__ _Float16 Vt[HD * VTP];           // 27 KB, [hd][seq] (transposed)
    __shared__ _Float16 Plds[8][16 * 32];       //  8 KB
    const int bh = blockIdx.x;
    const int b = bh / NHEAD, hh = bh % NHEAD;
    const int tid = threadIdx.x;
    const int lane = tid & 31, wid = tid >> 5;
    const int halfid = lane >> 4, col = lane & 15;
    const float scale = 0.125f;  // 1/sqrt(64)

    const float* qbase = qkv + (size_t)b * SEQLEN * (3 * DMODEL) + hh * HD;
    const float* kbase = qbase + DMODEL;
    const float* vbase = qbase + 2 * DMODEL;

    // stage K (row-major) and V (transposed), f32 -> f16, zero-pad rows >= SEQLEN
    for (int idx4 = tid; idx4 < (SEQ2 * HD) / 4; idx4 += 256) {
        int r = idx4 >> 4;          // seq row 0..207
        int c = (idx4 & 15) << 2;   // hd col 0,4,..,60
        float4 kv4 = make_float4(0.f, 0.f, 0.f, 0.f);
        float4 vv4 = make_float4(0.f, 0.f, 0.f, 0.f);
        if (r < SEQLEN) {
            kv4 = *(const float4*)&kbase[(size_t)r * (3 * DMODEL) + c];
            vv4 = *(const float4*)&vbase[(size_t)r * (3 * DMODEL) + c];
        }
        v4h kh; kh[0] = (_Float16)kv4.x; kh[1] = (_Float16)kv4.y;
        kh[2] = (_Float16)kv4.z; kh[3] = (_Float16)kv4.w;
        *(v4h*)&Klds[r * HD + c] = kh;
        Vt[(c + 0) * VTP + r] = (_Float16)vv4.x;
        Vt[(c + 1) * VTP + r] = (_Float16)vv4.y;
        Vt[(c + 2) * VTP + r] = (_Float16)vv4.z;
        Vt[(c + 3) * VTP + r] = (_Float16)vv4.w;
    }
    __syncthreads();

    for (int qt = wid; qt < 13; qt += 8) {   // 13 tiles of 16 query rows
        const int m0 = qt * 16;
        // Q A-fragments (K = 0..31, 32..63), scaled; 2x float4 per 8-group
        v16h qa[2];
        {
            int m = m0 + col;
            const float* qrow = qbase + (size_t)((m < SEQLEN) ? m : 0) * (3 * DMODEL);
            float zmask = (m < SEQLEN) ? scale : 0.f;
#pragma unroll
            for (int kb2 = 0; kb2 < 2; ++kb2) {
                const float* qp = qrow + kb2 * 32 + (halfid << 3);
                float4 q0 = *(const float4*)&qp[0];
                float4 q1 = *(const float4*)&qp[4];
                float4 q2 = *(const float4*)&qp[16];
                float4 q3 = *(const float4*)&qp[20];
                qa[kb2][0]  = (_Float16)(q0.x * zmask); qa[kb2][1]  = (_Float16)(q0.y * zmask);
                qa[kb2][2]  = (_Float16)(q0.z * zmask); qa[kb2][3]  = (_Float16)(q0.w * zmask);
                qa[kb2][4]  = (_Float16)(q1.x * zmask); qa[kb2][5]  = (_Float16)(q1.y * zmask);
                qa[kb2][6]  = (_Float16)(q1.z * zmask); qa[kb2][7]  = (_Float16)(q1.w * zmask);
                qa[kb2][8]  = (_Float16)(q2.x * zmask); qa[kb2][9]  = (_Float16)(q2.y * zmask);
                qa[kb2][10] = (_Float16)(q2.z * zmask); qa[kb2][11] = (_Float16)(q2.w * zmask);
                qa[kb2][12] = (_Float16)(q3.x * zmask); qa[kb2][13] = (_Float16)(q3.y * zmask);
                qa[kb2][14] = (_Float16)(q3.z * zmask); qa[kb2][15] = (_Float16)(q3.w * zmask);
            }
        }
        v8f oacc[4];
#pragma unroll
        for (int nf = 0; nf < 4; ++nf) oacc[nf] = v8f{};
        float rmax[8], rsumL[8];
#pragma unroll
        for (int r = 0; r < 8; ++r) { rmax[r] = -1e30f; rsumL[r] = 0.f; }

        for (int ct = 0; ct < 7; ++ct) {     // 7 steps of 32 key columns
            const int c0 = ct * 2;
            int kr0 = c0 * 16 + col;         // <= 207, always valid
            int kr1 = kr0 + 16;
            if (kr1 >= SEQ2) kr1 = 0;        // tile 13 is fully masked anyway
            v8f s0 = v8f{}, s1 = v8f{};
#pragma unroll
            for (int kb2 = 0; kb2 < 2; ++kb2) {
                const _Float16* kp0 = &Klds[kr0 * HD + kb2 * 32 + (halfid << 3)];
                const _Float16* kp1 = &Klds[kr1 * HD + kb2 * 32 + (halfid << 3)];
                v16h kf0 = frag_from_groups(*(const v8h*)kp0, *(const v8h*)&kp0[16]);
                v16h kf1 = frag_from_groups(*(const v8h*)kp1, *(const v8h*)&kp1[16]);
                s0 = __builtin_amdgcn_wmma_f32_16x16x32_f16(false, qa[kb2], false, kf0,
                                                            (short)0, s0, false, false);
                s1 = __builtin_amdgcn_wmma_f32_16x16x32_f16(false, qa[kb2], false, kf1,
                                                            (short)0, s1, false, false);
            }
            const int n0c = c0 * 16 + col;
            const int n1c = n0c + 16;
            // online softmax: rows map onto register index r (+8*halfid)
#pragma unroll
            for (int r = 0; r < 8; ++r) {
                float s0v = (n0c < SEQLEN) ? s0[r] : -1e30f;
                float s1v = (n1c < SEQLEN) ? s1[r] : -1e30f;
                float tmax = fmaxf(s0v, s1v);
#pragma unroll
                for (int o = 1; o < 16; o <<= 1) tmax = fmaxf(tmax, __shfl_xor(tmax, o, 32));
                float nm = fmaxf(rmax[r], tmax);
                float alpha = __expf(rmax[r] - nm);
                rmax[r] = nm;
                float p0 = __expf(s0v - nm);
                float p1 = __expf(s1v - nm);
                rsumL[r] = rsumL[r] * alpha + p0 + p1;   // lane-local partial sum
#pragma unroll
                for (int nf = 0; nf < 4; ++nf) oacc[nf][r] *= alpha;
                int prow = r + halfid * 8;
                Plds[wid][prow * 32 + col]      = (_Float16)p0;
                Plds[wid][prow * 32 + 16 + col] = (_Float16)p1;
            }
            // O += P[16x32] x V[32x64]; V fragments contiguous via transposed LDS
            const _Float16* pp = &Plds[wid][col * 32 + (halfid << 3)];
            v16h pa = frag_from_groups(*(const v8h*)pp, *(const v8h*)&pp[16]);
            int base0 = ct * 32 + (halfid << 3);   // <= 200, always valid
            int base1 = base0 + 16;
            if (base1 >= SEQ2) base1 = 0;          // P columns there are exactly 0
#pragma unroll
            for (int nf = 0; nf < 4; ++nf) {
                const _Float16* vp = &Vt[(nf * 16 + col) * VTP];
                v16h vf = frag_from_groups(*(const v8h*)&vp[base0], *(const v8h*)&vp[base1]);
                oacc[nf] = __builtin_amdgcn_wmma_f32_16x16x32_f16(false, pa, false, vf,
                                                                  (short)0, oacc[nf], false, false);
            }
        }
        // final cross-lane sum reduction, normalize & store
#pragma unroll
        for (int r = 0; r < 8; ++r) {
            float s = rsumL[r];
#pragma unroll
            for (int o = 1; o < 16; o <<= 1) s += __shfl_xor(s, o, 32);
            float inv = 1.f / s;
            int m = m0 + r + halfid * 8;
            if (m < SEQLEN) {
#pragma unroll
                for (int nf = 0; nf < 4; ++nf) {
                    out[((size_t)(b * SEQLEN + m)) * DMODEL + hh * HD + nf * 16 + col] =
                        oacc[nf][r] * inv;
                }
            }
        }
    }
}

// ---------------------------------------------------------------------------
// orchestration
// ---------------------------------------------------------------------------
extern "C" void kernel_launch(void* const* d_in, const int* in_sizes, int n_in,
                              void* d_out, int out_size, void* d_ws, size_t ws_size,
                              hipStream_t stream) {
    (void)in_sizes; (void)n_in; (void)out_size; (void)ws_size;
    const float* image   = (const float*)d_in[0];
    const float* conv_w  = (const float*)d_in[1];   // [768, 768]
    const float* class_e = (const float*)d_in[2];
    const float* pos_e   = (const float*)d_in[3];
    const float* lnpre_w = (const float*)d_in[4];
    const float* lnpre_b = (const float*)d_in[5];
    const float* ln1_w   = (const float*)d_in[6];
    const float* ln1_b   = (const float*)d_in[7];
    const float* qkv_w   = (const float*)d_in[8];
    const float* qkv_b   = (const float*)d_in[9];
    const float* out_w   = (const float*)d_in[10];
    const float* out_b   = (const float*)d_in[11];
    const float* ln2_w   = (const float*)d_in[12];
    const float* ln2_b   = (const float*)d_in[13];
    const float* fc_w    = (const float*)d_in[14];
    const float* fc_b    = (const float*)d_in[15];
    const float* proj_w  = (const float*)d_in[16];
    const float* proj_b  = (const float*)d_in[17];
    const float* lnpo_w  = (const float*)d_in[18];
    const float* lnpo_b  = (const float*)d_in[19];
    const float* head_w  = (const float*)d_in[20];
    const float* head_b  = (const float*)d_in[21];

    char* ws = (char*)d_ws;
    const size_t SZ_XD = (size_t)NTOK * DMODEL * sizeof(float);   // 38.7 MB
    float* X   = (float*)(ws);
    float* Hb  = (float*)(ws + SZ_XD);
    float* O   = (float*)(ws + 2 * SZ_XD);
    float* BIG = (float*)(ws + 3 * SZ_XD);                        // 155 MB (qkv / ffn)
    float* FE  = (float*)(ws + 3 * SZ_XD + (size_t)NTOK * FFDIM * sizeof(float));

    // patch embedding: im2col -> GEMM vs conv_w
    im2col_kernel<<<8192, 256, 0, stream>>>(image, Hb);
    {
        dim3 g(DMODEL / 128, PROWS / 128);
        gemm_kernel<0><<<g, 256, 0, stream>>>(Hb, DMODEL, conv_w, DMODEL, nullptr, nullptr,
                                              O, DMODEL, PROWS, DMODEL, DMODEL);
    }
    assemble_ln_kernel<<<NTOK, 256, 0, stream>>>(O, class_e, pos_e, lnpre_w, lnpre_b, X);

    const int GY = (NTOK + 127) / 128;  // 99
    for (int l = 0; l < NLAYER; ++l) {
        ln_kernel<<<NTOK, 256, 0, stream>>>(X, ln1_w + l * DMODEL, ln1_b + l * DMODEL, Hb, DMODEL);
        {
            dim3 g((3 * DMODEL) / 128, GY);
            gemm_kernel<0><<<g, 256, 0, stream>>>(Hb, DMODEL,
                                                  qkv_w + (size_t)l * 3 * DMODEL * DMODEL, DMODEL,
                                                  qkv_b + (size_t)l * 3 * DMODEL, nullptr,
                                                  BIG, 3 * DMODEL, NTOK, 3 * DMODEL, DMODEL);
        }
        attn_kernel<<<NBATCH * NHEAD, 256, 0, stream>>>(BIG, O);
        {
            dim3 g(DMODEL / 128, GY);
            gemm_kernel<1><<<g, 256, 0, stream>>>(O, DMODEL,
                                                  out_w + (size_t)l * DMODEL * DMODEL, DMODEL,
                                                  out_b + (size_t)l * DMODEL, X,
                                                  X, DMODEL, NTOK, DMODEL, DMODEL);
        }
        ln_kernel<<<NTOK, 256, 0, stream>>>(X, ln2_w + l * DMODEL, ln2_b + l * DMODEL, Hb, DMODEL);
        {
            dim3 g(FFDIM / 128, GY);
            gemm_kernel<2><<<g, 256, 0, stream>>>(Hb, DMODEL,
                                                  fc_w + (size_t)l * FFDIM * DMODEL, DMODEL,
                                                  fc_b + (size_t)l * FFDIM, nullptr,
                                                  BIG, FFDIM, NTOK, FFDIM, DMODEL);
        }
        {
            dim3 g(DMODEL / 128, GY);
            gemm_kernel<1><<<g, 256, 0, stream>>>(BIG, FFDIM,
                                                  proj_w + (size_t)l * DMODEL * FFDIM, FFDIM,
                                                  proj_b + (size_t)l * DMODEL, X,
                                                  X, DMODEL, NTOK, DMODEL, FFDIM);
        }
    }

    // final LN on cls rows + classifier head
    ln_kernel<<<NBATCH, 256, 0, stream>>>(X, lnpo_w, lnpo_b, FE, SEQLEN * DMODEL);
    {
        dim3 g((1000 + 127) / 128, 1);
        gemm_kernel<0><<<g, 256, 0, stream>>>(FE, DMODEL, head_w, DMODEL, head_b, nullptr,
                                              (float*)d_out, 1000, NBATCH, 1000, DMODEL);
    }
}
